// ShiTomasiSparseBADSinkhornMatcher_26577257628024
// MI455X (gfx1250) — compile-verified, matrix operand order checked by reading, not run on hardware
//
#include <hip/hip_runtime.h>
#include <hip/hip_bf16.h>
#include <math.h>

#define BATCH 8
#define HH 720
#define WW 1280
#define PD 256
#define KK 1024
#define KP1 1025
#define CAP 32768

typedef __attribute__((ext_vector_type(16))) _Float16 v16h;
typedef __attribute__((ext_vector_type(8)))  _Float16 v8h;
typedef __attribute__((ext_vector_type(8)))  float    v8f;

// ---------------------------------------------------------------------------
// Phase 1: Shi-Tomasi score (fused Sobel + 3x3 box + min-eigenvalue)
// ---------------------------------------------------------------------------
__global__ void shi_tomasi_kernel(const float* __restrict__ img, float* __restrict__ scores) {
  long gid = (long)blockIdx.x * blockDim.x + threadIdx.x;
  const long total = (long)BATCH * HH * WW;
  if (gid >= total) return;
  int x = (int)(gid % WW);
  int y = (int)((gid / WW) % HH);
  int b = (int)(gid / ((long)HH * WW));
  const float* im = img + (size_t)b * HH * WW;

  float p[5][5];
  #pragma unroll
  for (int i = 0; i < 5; ++i)
    #pragma unroll
    for (int j = 0; j < 5; ++j) {
      int yy = y + i - 2, xx = x + j - 2;
      p[i][j] = (yy >= 0 && yy < HH && xx >= 0 && xx < WW) ? im[(size_t)yy * WW + xx] : 0.0f;
    }

  float sxx = 0.f, syy = 0.f, sxy = 0.f;
  #pragma unroll
  for (int cy = -1; cy <= 1; ++cy)
    #pragma unroll
    for (int cx = -1; cx <= 1; ++cx) {
      int yy = y + cy, xx = x + cx;
      if (yy < 0 || yy >= HH || xx < 0 || xx >= WW) continue;  // box conv zero-pads ix*ix
      int r = 2 + cy, c = 2 + cx;
      float gx = (p[r-1][c+1] - p[r-1][c-1]) + 2.f*(p[r][c+1] - p[r][c-1]) + (p[r+1][c+1] - p[r+1][c-1]);
      float gy = (p[r+1][c-1] - p[r-1][c-1]) + 2.f*(p[r+1][c] - p[r-1][c]) + (p[r+1][c+1] - p[r-1][c+1]);
      sxx += gx * gx; syy += gy * gy; sxy += gx * gy;
    }
  sxx *= (1.f/9.f); syy *= (1.f/9.f); sxy *= (1.f/9.f);
  float tr  = 0.5f * (sxx + syy);
  float hd  = 0.5f * (sxx - syy);
  float rad = sqrtf(hd * hd + sxy * sxy + 1e-12f);
  scores[gid] = tr - rad;
}

// ---------------------------------------------------------------------------
// Phase 2: 7x7 NMS -> sparse candidate list (atomic append)
// ---------------------------------------------------------------------------
__global__ void nms_kernel(const float* __restrict__ scores,
                           float* __restrict__ cand_val, int* __restrict__ cand_idx,
                           int* __restrict__ cand_cnt) {
  long gid = (long)blockIdx.x * blockDim.x + threadIdx.x;
  const long total = (long)BATCH * HH * WW;
  if (gid >= total) return;
  int x = (int)(gid % WW);
  int y = (int)((gid / WW) % HH);
  int b = (int)(gid / ((long)HH * WW));
  float s = scores[gid];
  if (s <= 0.f) return;
  const float* sc = scores + (size_t)b * HH * WW;
  float lm = -INFINITY;
  for (int dy = -3; dy <= 3; ++dy) {
    int yy = y + dy; if (yy < 0 || yy >= HH) continue;
    for (int dx = -3; dx <= 3; ++dx) {
      int xx = x + dx; if (xx < 0 || xx >= WW) continue;
      lm = fmaxf(lm, sc[(size_t)yy * WW + xx]);
    }
  }
  if (s >= lm - 1e-7f) {
    int pos = atomicAdd(&cand_cnt[b], 1);
    if (pos < CAP) {
      cand_val[(size_t)b * CAP + pos] = s;
      cand_idx[(size_t)b * CAP + pos] = y * WW + x;
    }
  }
}

// ---------------------------------------------------------------------------
// Phase 3: exact top-K=1024 per (image,batch) via repeated block argmax.
// Ties broken by lowest pixel index (matches jax.lax.top_k); deterministic
// regardless of atomic append order.
// ---------------------------------------------------------------------------
__global__ void __launch_bounds__(1024) topk_kernel(float* __restrict__ cand_val,
                                                    const int* __restrict__ cand_idx,
                                                    const int* __restrict__ cand_cnt,
                                                    float* __restrict__ sel_val,
                                                    int* __restrict__ sel_idx) {
  int bb = blockIdx.x;                    // img*8 + b, 0..15
  int n = cand_cnt[bb]; if (n > CAP) n = CAP;
  float* vals = cand_val + (size_t)bb * CAP;
  const int* idxs = cand_idx + (size_t)bb * CAP;
  __shared__ float sv[1024];
  __shared__ int   si[1024];
  __shared__ int   ss[1024];
  int tid = threadIdx.x;
  for (int k = 0; k < KK; ++k) {
    float bv = -INFINITY; int bi = 0x7fffffff; int bs = -1;
    for (int slot = tid; slot < n; slot += 1024) {
      float vv = vals[slot]; int id = idxs[slot];
      if (vv > bv || (vv == bv && id < bi)) { bv = vv; bi = id; bs = slot; }
    }
    sv[tid] = bv; si[tid] = bi; ss[tid] = bs;
    __syncthreads();
    for (int off = 512; off > 0; off >>= 1) {
      if (tid < off) {
        float ov = sv[tid + off]; int oi = si[tid + off];
        if (ov > sv[tid] || (ov == sv[tid] && oi < si[tid])) {
          sv[tid] = ov; si[tid] = oi; ss[tid] = ss[tid + off];
        }
      }
      __syncthreads();
    }
    if (tid == 0) {
      if (ss[0] >= 0 && sv[0] > 0.f) {
        sel_val[(size_t)bb * KK + k] = sv[0];
        sel_idx[(size_t)bb * KK + k] = si[0];
        vals[ss[0]] = -INFINITY;            // remove from pool
      } else {
        sel_val[(size_t)bb * KK + k] = 0.f; // invalid slot
        sel_idx[(size_t)bb * KK + k] = 0;
      }
    }
    __syncthreads();
  }
}

// ---------------------------------------------------------------------------
// Phase 4: BAD descriptors. Box bank computed on the fly with edge clamping
// (== pad 'edge' + VALID conv). One block per keypoint, thread = descriptor dim.
// Normalized output stored as f16 (WMMA input) + exact fp32 squared norm.
// ---------------------------------------------------------------------------
__device__ inline float box_sample(const float* __restrict__ im, float yc, float xc,
                                   float oy, float ox, int r) {
  int yi = (int)rintf(yc + oy); yi = yi < 0 ? 0 : (yi > HH - 1 ? HH - 1 : yi);
  int xi = (int)rintf(xc + ox); xi = xi < 0 ? 0 : (xi > WW - 1 ? WW - 1 : xi);
  float acc = 0.f;
  for (int dy = -r; dy <= r; ++dy) {
    int yy = yi + dy; yy = yy < 0 ? 0 : (yy > HH - 1 ? HH - 1 : yy);
    for (int dx = -r; dx <= r; ++dx) {
      int xx = xi + dx; xx = xx < 0 ? 0 : (xx > WW - 1 ? WW - 1 : xx);
      acc += im[(size_t)yy * WW + xx];
    }
  }
  float side = (float)(2 * r + 1);
  return acc / (side * side);
}

__global__ void __launch_bounds__(PD) desc_kernel(const float* __restrict__ img,
                            const float* __restrict__ sel_val, const int* __restrict__ sel_idx,
                            const float* __restrict__ offsets, const float* __restrict__ thresholds,
                            const int* __restrict__ radii,
                            _Float16* __restrict__ desc_h, float* __restrict__ nsq) {
  int bk  = blockIdx.x;                 // b*KK + k
  int b   = bk / KK;
  int tid = threadIdx.x;
  float val = sel_val[bk];
  int   id  = sel_idx[bk];
  bool valid = val > 0.f;
  float yc = (float)(id / WW), xc = (float)(id % WW);
  const float* im = img + (size_t)b * HH * WW;

  float d = 0.f;
  if (valid) {
    float ox1 = offsets[0 * PD + tid], ox2 = offsets[1 * PD + tid];
    float oy1 = offsets[2 * PD + tid], oy2 = offsets[3 * PD + tid];
    int r = radii[tid]; r = r < 0 ? 0 : (r > 3 ? 3 : r);
    float s1 = box_sample(im, yc, xc, oy1, ox1, r);
    float s2 = box_sample(im, yc, xc, oy2, ox2, r);
    d = s1 - s2 - thresholds[tid];
  }
  __shared__ float red[PD];
  red[tid] = d * d;
  __syncthreads();
  for (int off = PD / 2; off > 0; off >>= 1) {
    if (tid < off) red[tid] += red[tid + off];
    __syncthreads();
  }
  float sum = red[0];
  float nrm = sqrtf(sum) + 1e-8f;
  float o = d / nrm;
  desc_h[(size_t)bk * PD + tid] = (_Float16)o;
  if (tid == 0) nsq[bk] = sum / (nrm * nrm);
}

// ---------------------------------------------------------------------------
// Phase 5: cost GEMM via v_wmma_f32_16x16x32_f16.
// D[i,j] = d1_i . d2_j  (A row-major per ISA A-layout, B mirrored with N on lanes)
// cost = sqrt(max(n1+n2-2*dot,0)+1e-12)
// ---------------------------------------------------------------------------
__device__ inline v16h load_frag(const _Float16* __restrict__ base, int row, int kc, int hi) {
  // lane<16 (hi=0): halves = K {kc..kc+7, kc+16..kc+23}
  // lane>=16 (hi=1): halves = K {kc+8..kc+15, kc+24..kc+31}
  const _Float16* p = base + (size_t)row * PD + kc + hi * 8;
  v8h lo = *(const v8h*)(p);
  v8h hh = *(const v8h*)(p + 16);
  v16h f;
  #pragma unroll
  for (int t = 0; t < 8; ++t) { f[t] = lo[t]; f[t + 8] = hh[t]; }
  return f;
}

__global__ void __launch_bounds__(128) cost_gemm_kernel(const _Float16* __restrict__ dA,
                                 const _Float16* __restrict__ dB,
                                 const float* __restrict__ nA, const float* __restrict__ nB,
                                 float* __restrict__ cost) {
  int b    = blockIdx.z;
  int wave = threadIdx.x >> 5;
  int lane = threadIdx.x & 31;
  int wm = blockIdx.y * 64 + (wave >> 1) * 32;   // rows (d1 keypoints)
  int wn = blockIdx.x * 64 + (wave & 1) * 32;    // cols (d2 keypoints)
  int rr = lane & 15, hi = lane >> 4;
  const _Float16* aB = dA + (size_t)b * KK * PD;
  const _Float16* bB = dB + (size_t)b * KK * PD;

  v8f acc[2][2] = {};
  for (int kc = 0; kc < PD; kc += 32) {
    v16h af[2], bf[2];
    #pragma unroll
    for (int t = 0; t < 2; ++t) {
      af[t] = load_frag(aB, wm + t * 16 + rr, kc, hi);
      bf[t] = load_frag(bB, wn + t * 16 + rr, kc, hi);
    }
    #pragma unroll
    for (int i = 0; i < 2; ++i)
      #pragma unroll
      for (int j = 0; j < 2; ++j)
        acc[i][j] = __builtin_amdgcn_wmma_f32_16x16x32_f16(
            false, af[i], false, bf[j], (short)0, acc[i][j], false, false);
  }

  // C/D layout: element e -> M = e + hi*8 ; N = lane&15
  #pragma unroll
  for (int ti = 0; ti < 2; ++ti)
    #pragma unroll
    for (int tj = 0; tj < 2; ++tj) {
      int j = wn + tj * 16 + rr;
      float n2 = nB[(size_t)b * KK + j];
      #pragma unroll
      for (int e = 0; e < 8; ++e) {
        int i = wm + ti * 16 + hi * 8 + e;
        float sq = nA[(size_t)b * KK + i] + n2 - 2.f * acc[ti][tj][e];
        sq = fmaxf(sq, 0.f);
        cost[((size_t)b * KK + i) * KK + j] = sqrtf(sq + 1e-12f);
      }
    }
}

// ---------------------------------------------------------------------------
// Phase 6: Sinkhorn log-domain updates. Z[:K,:K] = -cost, dust row/col = 1.0
// (epsilon = 1, unused_score = 1). log_mu[i<K] = -log(2K), log_mu[K] = -log 2.
// ---------------------------------------------------------------------------
__global__ void __launch_bounds__(256) sinkhorn_u_kernel(const float* __restrict__ cost,
                                  const float* __restrict__ v, float* __restrict__ u) {
  int i = blockIdx.x, b = blockIdx.y, tid = threadIdx.x;
  const float* vb = v + (size_t)b * KP1;
  float t[5]; float m = -INFINITY;
  #pragma unroll
  for (int s = 0; s < 5; ++s) {
    int j = tid + s * 256;
    float term = -INFINITY;
    if (j < KP1) {
      float Z = (i < KK && j < KK) ? -cost[((size_t)b * KK + i) * KK + j] : 1.0f;
      term = Z + vb[j];
    }
    t[s] = term; m = fmaxf(m, term);
  }
  __shared__ float red[256];
  red[tid] = m; __syncthreads();
  for (int off = 128; off > 0; off >>= 1) { if (tid < off) red[tid] = fmaxf(red[tid], red[tid + off]); __syncthreads(); }
  float M = red[0]; __syncthreads();
  float ssum = 0.f;
  #pragma unroll
  for (int s = 0; s < 5; ++s) ssum += (t[s] == -INFINITY) ? 0.f : expf(t[s] - M);
  red[tid] = ssum; __syncthreads();
  for (int off = 128; off > 0; off >>= 1) { if (tid < off) red[tid] += red[tid + off]; __syncthreads(); }
  if (tid == 0) {
    float lmu = (i < KK) ? -logf(2048.0f) : -logf(2.0f);
    u[(size_t)b * KP1 + i] = lmu - (M + logf(red[0]));
  }
}

__global__ void __launch_bounds__(256) sinkhorn_v_kernel(const float* __restrict__ cost,
                                  const float* __restrict__ u, float* __restrict__ v) {
  int j = blockIdx.x, b = blockIdx.y, tid = threadIdx.x;
  const float* ub = u + (size_t)b * KP1;
  float t[5]; float m = -INFINITY;
  #pragma unroll
  for (int s = 0; s < 5; ++s) {
    int i = tid + s * 256;
    float term = -INFINITY;
    if (i < KP1) {
      float Z = (i < KK && j < KK) ? -cost[((size_t)b * KK + i) * KK + j] : 1.0f;
      term = Z + ub[i];
    }
    t[s] = term; m = fmaxf(m, term);
  }
  __shared__ float red[256];
  red[tid] = m; __syncthreads();
  for (int off = 128; off > 0; off >>= 1) { if (tid < off) red[tid] = fmaxf(red[tid], red[tid + off]); __syncthreads(); }
  float M = red[0]; __syncthreads();
  float ssum = 0.f;
  #pragma unroll
  for (int s = 0; s < 5; ++s) ssum += (t[s] == -INFINITY) ? 0.f : expf(t[s] - M);
  red[tid] = ssum; __syncthreads();
  for (int off = 128; off > 0; off >>= 1) { if (tid < off) red[tid] += red[tid + off]; __syncthreads(); }
  if (tid == 0) {
    float lnu = (j < KK) ? -logf(2048.0f) : -logf(2.0f);
    v[(size_t)b * KP1 + j] = lnu - (M + logf(red[0]));
  }
}

__global__ void probs_kernel(const float* __restrict__ cost, const float* __restrict__ u,
                             const float* __restrict__ v, float* __restrict__ out) {
  long gid = (long)blockIdx.x * blockDim.x + threadIdx.x;
  const long total = (long)BATCH * KP1 * KP1;
  if (gid >= total) return;
  int j = (int)(gid % KP1);
  int i = (int)((gid / KP1) % KP1);
  int b = (int)(gid / ((long)KP1 * KP1));
  float Z = (i < KK && j < KK) ? -cost[((size_t)b * KK + i) * KK + j] : 1.0f;
  out[gid] = expf(Z + u[(size_t)b * KP1 + i] + v[(size_t)b * KP1 + j]);
}

__global__ void kpts_kernel(const float* __restrict__ sel_val, const int* __restrict__ sel_idx,
                            float* __restrict__ out) {
  int gid = blockIdx.x * blockDim.x + threadIdx.x;   // [img][b][k] flat, 0..16383
  if (gid >= 2 * BATCH * KK) return;
  float val = sel_val[gid]; int id = sel_idx[gid];
  float y = -1.f, x = -1.f;
  if (val > 0.f) { y = (float)(id / WW); x = (float)(id % WW); }
  out[(size_t)gid * 2 + 0] = y;
  out[(size_t)gid * 2 + 1] = x;
}

// ---------------------------------------------------------------------------
// Host launch
// ---------------------------------------------------------------------------
extern "C" void kernel_launch(void* const* d_in, const int* in_sizes, int n_in,
                              void* d_out, int out_size, void* d_ws, size_t ws_size,
                              hipStream_t stream) {
  (void)in_sizes; (void)n_in; (void)out_size; (void)ws_size;
  const float* img1       = (const float*)d_in[0];
  const float* img2       = (const float*)d_in[1];
  const float* offsets    = (const float*)d_in[2];
  const float* thresholds = (const float*)d_in[3];
  const int*   radii      = (const int*)d_in[4];
  // d_in[5] = max_keypoints = 1024 (compile-time constant KK)

  char* ws = (char*)d_ws;
  size_t off = 0;
  auto alloc = [&](size_t bytes) { size_t o = off; off = (off + bytes + 255) & ~(size_t)255; return o; };
  size_t o_sc1 = alloc((size_t)BATCH * HH * WW * 4);
  size_t o_sc2 = alloc((size_t)BATCH * HH * WW * 4);
  size_t o_cv  = alloc((size_t)2 * BATCH * CAP * 4);
  size_t o_ci  = alloc((size_t)2 * BATCH * CAP * 4);
  size_t o_cc  = alloc((size_t)2 * BATCH * 4);
  size_t o_sv  = alloc((size_t)2 * BATCH * KK * 4);
  size_t o_si  = alloc((size_t)2 * BATCH * KK * 4);
  size_t o_dh  = alloc((size_t)2 * BATCH * KK * PD * 2);
  size_t o_n   = alloc((size_t)2 * BATCH * KK * 4);
  size_t o_cost= alloc((size_t)BATCH * KK * KK * 4);
  size_t o_u   = alloc((size_t)BATCH * KP1 * 4);
  size_t o_v   = alloc((size_t)BATCH * KP1 * 4);

  float*     sc1 = (float*)(ws + o_sc1);
  float*     sc2 = (float*)(ws + o_sc2);
  float*     cv  = (float*)(ws + o_cv);
  int*       ci  = (int*)(ws + o_ci);
  int*       cc  = (int*)(ws + o_cc);
  float*     sv  = (float*)(ws + o_sv);
  int*       si  = (int*)(ws + o_si);
  _Float16*  dh  = (_Float16*)(ws + o_dh);
  float*     nn  = (float*)(ws + o_n);
  float*     cost= (float*)(ws + o_cost);
  float*     uu  = (float*)(ws + o_u);
  float*     vv  = (float*)(ws + o_v);

  hipMemsetAsync(ws + o_cc, 0, (size_t)2 * BATCH * 4, stream);

  const long npix = (long)BATCH * HH * WW;
  const int  TB = 256;
  const long NB = (npix + TB - 1) / TB;

  shi_tomasi_kernel<<<NB, TB, 0, stream>>>(img1, sc1);
  shi_tomasi_kernel<<<NB, TB, 0, stream>>>(img2, sc2);

  nms_kernel<<<NB, TB, 0, stream>>>(sc1, cv, ci, cc);
  nms_kernel<<<NB, TB, 0, stream>>>(sc2, cv + (size_t)BATCH * CAP, ci + (size_t)BATCH * CAP, cc + BATCH);

  topk_kernel<<<2 * BATCH, 1024, 0, stream>>>(cv, ci, cc, sv, si);

  desc_kernel<<<BATCH * KK, PD, 0, stream>>>(img1, sv, si, offsets, thresholds, radii, dh, nn);
  desc_kernel<<<BATCH * KK, PD, 0, stream>>>(img2, sv + (size_t)BATCH * KK, si + (size_t)BATCH * KK,
                                             offsets, thresholds, radii,
                                             dh + (size_t)BATCH * KK * PD, nn + (size_t)BATCH * KK);

  dim3 ggrid(KK / 64, KK / 64, BATCH);
  cost_gemm_kernel<<<ggrid, 128, 0, stream>>>(dh, dh + (size_t)BATCH * KK * PD,
                                              nn, nn + (size_t)BATCH * KK, cost);

  hipMemsetAsync(ws + o_u, 0, (size_t)BATCH * KP1 * 4, stream);
  hipMemsetAsync(ws + o_v, 0, (size_t)BATCH * KP1 * 4, stream);

  dim3 sgrid(KP1, BATCH);
  for (int it = 0; it < 20; ++it) {
    sinkhorn_u_kernel<<<sgrid, 256, 0, stream>>>(cost, vv, uu);
    sinkhorn_v_kernel<<<sgrid, 256, 0, stream>>>(cost, uu, vv);
  }

  float* out = (float*)d_out;
  const long nprob = (long)BATCH * KP1 * KP1;
  probs_kernel<<<(nprob + 255) / 256, 256, 0, stream>>>(cost, uu, vv, out + (size_t)2 * 2 * BATCH * KK);
  kpts_kernel<<<(2 * BATCH * KK + 255) / 256, 256, 0, stream>>>(sv, si, out);
}
